// FP8LinearModel_20005957664793
// MI455X (gfx1250) — 8-bit matrix operand order repaired and checked by reading, compiled, not run on hardware
//
#include <hip/hip_runtime.h>


typedef int   v16i __attribute__((ext_vector_type(16)));
typedef int   v4i  __attribute__((ext_vector_type(4)));
typedef int   v2i  __attribute__((ext_vector_type(2)));
typedef float v8f  __attribute__((ext_vector_type(8)));
typedef float v4f  __attribute__((ext_vector_type(4)));

#define FP8_MAX 448.0f


static __device__ __forceinline__ unsigned char sw_f32_to_e4m3(float f) {
  union { float f; unsigned u; } v; v.f = f;
  unsigned sgn = (v.u >> 31) << 7;
  unsigned au  = v.u & 0x7fffffffu;
  if (au == 0) return (unsigned char)sgn;
  int e = (int)(au >> 23) - 127;
  unsigned m = au & 0x7fffffu;
  if (e < -6) {
    union { float f; unsigned u; } a; a.u = au;
    int qi = (int)(a.f * 512.0f + 0.5f);
    if (qi >= 8) return (unsigned char)(sgn | 0x08);
    return (unsigned char)(sgn | (unsigned)qi);
  }
  unsigned r = m + 0x7ffffu + ((m >> 20) & 1u);
  if (r >= 0x800000u) { e += 1; r = 0; }
  unsigned m3 = (r >> 20) & 7u;
  if (e > 8) { e = 8; m3 = 6; }
  return (unsigned char)(sgn | ((unsigned)(e + 7) << 3) | m3);
}

static __device__ __forceinline__ unsigned pack4_fp8(float x0, float x1,
                                                     float x2, float x3) {
#if __has_builtin(__builtin_amdgcn_cvt_pk_fp8_f32)
  int r = __builtin_amdgcn_cvt_pk_fp8_f32(x0, x1, 0, false);
  r     = __builtin_amdgcn_cvt_pk_fp8_f32(x2, x3, r, true);
  return (unsigned)r;
#else
  return (unsigned)sw_f32_to_e4m3(x0)         |
         ((unsigned)sw_f32_to_e4m3(x1) << 8)  |
         ((unsigned)sw_f32_to_e4m3(x2) << 16) |
         ((unsigned)sw_f32_to_e4m3(x3) << 24);
#endif
}


__global__ __launch_bounds__(256) void quant_fp8_kernel(
    const float* __restrict__ src, unsigned* __restrict__ dst,
    const float* __restrict__ scale, int n4) {
  int i = blockIdx.x * blockDim.x + threadIdx.x;
  if (i >= n4) return;
  float inv = scale ? (1.0f / scale[0]) : 1.0f;
  v4f v = __builtin_nontemporal_load((const v4f*)src + i);
  float a = fminf(fmaxf(v.x * inv, -FP8_MAX), FP8_MAX);
  float b = fminf(fmaxf(v.y * inv, -FP8_MAX), FP8_MAX);
  float c = fminf(fmaxf(v.z * inv, -FP8_MAX), FP8_MAX);
  float d = fminf(fmaxf(v.w * inv, -FP8_MAX), FP8_MAX);
  dst[i] = pack4_fp8(a, b, c, d);
}


static __device__ __forceinline__ void load_a_frag(v16i& A,
                                                   const unsigned char* p) {
#pragma unroll
  for (int j = 0; j < 8; ++j) {
    v2i t = *(const v2i*)(p + 32 * (j >> 1) + 8 * (j & 1));
    A[2 * j] = t.x; A[2 * j + 1] = t.y;
  }
}

static __device__ __forceinline__ void load_b_frag(v16i& B,
                                                   const unsigned char* p) {
#pragma unroll
  for (int q = 0; q < 4; ++q) {
    v4i t = *(const v4i*)(p + 32 * q);
    B[4 * q]     = t.x; B[4 * q + 1] = t.y;
    B[4 * q + 2] = t.z; B[4 * q + 3] = t.w;
  }
}

__global__ __launch_bounds__(256) void fp8_gemm_bias_kernel(
    const unsigned char* __restrict__ a8,
    const unsigned char* __restrict__ w8,
    const float* __restrict__ bias,
    const float* __restrict__ in_scale,
    const float* __restrict__ w_scale,
    float* __restrict__ out,
    int M, int N, int K) {
  const int lane  = threadIdx.x & 31;
  const int wave  = threadIdx.x >> 5;
  const int wm    = wave & 3;
  const int wn    = wave >> 2;
  const int m0    = blockIdx.y * 128 + wm * 32;
  const int n0    = blockIdx.x * 128 + wn * 64;
  const int lhalf = lane >> 4;
  const int l15   = lane & 15;

  v8f acc[2][4];
#pragma unroll
  for (int i = 0; i < 2; ++i)
#pragma unroll
    for (int f = 0; f < 4; ++f)
      acc[i][f] = (v8f)0.0f;

  const unsigned char* arow0 = a8 + (size_t)(m0 + l15) * K + (size_t)(lhalf * 16);
  const unsigned char* arow1 = arow0 + (size_t)16 * K;
  const unsigned char* brow[4];
#pragma unroll
  for (int f = 0; f < 4; ++f)
    brow[f] = w8 + (size_t)(n0 + 16 * f + l15) * K + (size_t)(lhalf * 16);

  v16i Bbuf[2];
  load_b_frag(Bbuf[0], brow[0]);

  for (int kb = 0; kb < K; kb += 128) {
    const int kbn = (kb + 128 < K) ? (kb + 128) : kb;

    v16i A0, A1;
    load_a_frag(A0, arow0 + kb);
    load_a_frag(A1, arow1 + kb);

#pragma unroll
    for (int f = 0; f < 4; ++f) {
      const int cur = f & 1;
      const int nxt = cur ^ 1;
      if (f < 3) load_b_frag(Bbuf[nxt], brow[f + 1] + kb);
      else       load_b_frag(Bbuf[nxt], brow[0] + kbn);
      acc[0][f] = __builtin_amdgcn_wmma_f32_16x16x128_fp8_fp8(
          A0, Bbuf[cur], (short)0, acc[0][f], false, false);
      acc[1][f] = __builtin_amdgcn_wmma_f32_16x16x128_fp8_fp8(
          A1, Bbuf[cur], (short)0, acc[1][f], false, false);
    }
  }

  const float s = in_scale[0] * w_scale[0];
#pragma unroll
  for (int f = 0; f < 4; ++f) {
    const int n   = n0 + 16 * f + l15;
    const float bv = bias[n];
#pragma unroll
    for (int i = 0; i < 2; ++i) {
      const int mbase = m0 + 16 * i + lhalf * 8;
      float* po = out + (size_t)mbase * N + n;
#pragma unroll
      for (int r = 0; r < 8; ++r)
        __builtin_nontemporal_store(acc[i][f][r] * s + bv, po + (size_t)r * N);
    }
  }
}


extern "C" void kernel_launch(void* const* d_in, const int* in_sizes, int n_in,
                              void* d_out, int out_size, void* d_ws,
                              size_t ws_size, hipStream_t stream) {
  const float* x        = (const float*)d_in[0];
  const float* w        = (const float*)d_in[1];
  const float* bias     = (const float*)d_in[2];
  const float* in_scale = (const float*)d_in[3];
  const float* w_scale  = (const float*)d_in[4];
  float* out            = (float*)d_out;

  const int N = in_sizes[2];
  const int K = in_sizes[1] / N;
  const int M = in_sizes[0] / K;

  unsigned char* x8 = (unsigned char*)d_ws;
  unsigned char* w8 = x8 + (size_t)M * (size_t)K;

  const int xn4 = (M * K) / 4;
  quant_fp8_kernel<<<(xn4 + 255) / 256, 256, 0, stream>>>(
      x, (unsigned*)x8, in_scale, xn4);
  const int wn4 = (N * K) / 4;
  quant_fp8_kernel<<<(wn4 + 255) / 256, 256, 0, stream>>>(
      w, (unsigned*)w8, nullptr, wn4);

  dim3 grid(N / 128, M / 128);
  fp8_gemm_bias_kernel<<<grid, 256, 0, stream>>>(
      x8, w8, bias, in_scale, w_scale, out, M, N, K);
}
